// CausalSelfAttention_51840255262843
// MI455X (gfx1250) — compile-verified, matrix operand order checked by reading, not run on hardware
//
#include <hip/hip_runtime.h>
#include <hip/hip_bf16.h>

// ---------------------------------------------------------------------------
// CDNA5 (gfx1250) causal self-attention: qkv GEMM -> flash attention -> proj
// Matmuls via v_wmma_f32_16x16x32_f16 (wave32); GEMM tiles staged with the
// Tensor Data Mover (tensor_load_to_lds + s_wait_tensorcnt), double-buffered;
// attention K/V tiles software-prefetched (global_prefetch_b8).
// ---------------------------------------------------------------------------

typedef _Float16     v16h __attribute__((ext_vector_type(16)));
typedef _Float16     v8h  __attribute__((ext_vector_type(8)));
typedef float        v8f  __attribute__((ext_vector_type(8)));
typedef unsigned int v4u  __attribute__((ext_vector_type(4)));
typedef int          v8i  __attribute__((ext_vector_type(8)));
typedef int          v4i  __attribute__((ext_vector_type(4)));

#define D_MODEL 1024
#define N_HEAD  16
#define D_HEAD  64
#define SEQ_T   2048
#define BATCH   4
#define MROWS   (BATCH * SEQ_T)   // 8192

#if defined(__has_builtin)
#if __has_builtin(__builtin_amdgcn_tensor_load_to_lds) && \
    __has_builtin(__builtin_amdgcn_s_wait_tensorcnt)
#define USE_TDM 1
#endif
#endif
#ifndef USE_TDM
#define USE_TDM 0
#endif

__device__ __forceinline__ v8f wmma_f16(v16h a, v16h b, v8f c) {
  // (neg_a, A, neg_b, B, c_mod, C, reuse_a, reuse_b)
  return __builtin_amdgcn_wmma_f32_16x16x32_f16(false, a, false, b, (short)0, c,
                                                false, false);
}

// Load a 16x32 f16 fragment (A-layout; B uses the same striping when the
// source is stored [N][K] row-major). `stride` in halves, multiple of 8.
__device__ __forceinline__ v16h load_frag16(const _Float16* base, int stride) {
  const int lane = threadIdx.x & 31;
  const int row  = lane & 15;
  const int kc   = (lane < 16) ? 0 : 8;   // halves 0-7: K=kc..kc+7
  const _Float16* p = base + row * stride + kc;
  v8h lo = *(const v8h*)(p);              // K = kc .. kc+7
  v8h hi = *(const v8h*)(p + 16);         // K = kc+16 .. kc+23
  v16h r;
#pragma unroll
  for (int i = 0; i < 8; ++i) { r[i] = lo[i]; r[8 + i] = hi[i]; }
  return r;
}

#if USE_TDM
// Issue a TDM 2D tile load: tile_h rows x 32 halves, row stride = strideElems
// halves in global memory. The TDM itself inserts 16B of LDS padding per 64B
// row (pad_interval code 3 = every 16 DWORDs, pad_amount code 3 = 4 DWORDs),
// producing the 40-half LDS row stride the WMMA fragment loader expects.
__device__ __forceinline__ void tdm_load_tile32(const _Float16* gsrc,
                                                unsigned ldsByteAddr,
                                                unsigned tileRows,
                                                unsigned strideElems) {
  unsigned long long ga = (unsigned long long)gsrc;
  v4u g0;
  g0[0] = 1u;                                   // count=1, user mode, no gather
  g0[1] = ldsByteAddr;                          // lds_addr
  g0[2] = (unsigned)(ga & 0xffffffffu);         // global_addr[31:0]
  g0[3] = (unsigned)((ga >> 32) & 0x01ffffffu)  // global_addr[56:32]
          | (2u << 30);                         // type=2 ("image")
  const unsigned td0 = 0x7fffffffu;             // tensor_dim0 (no OOB clip)
  const unsigned td1 = 0x7fffffffu;             // tensor_dim1
  v8i g1;
  g1[0] = (int)((1u << 16)        // data_size=1 -> 2-byte elements
                | (1u << 20)      // pad_enable
                | (3u << 22)      // pad_interval: every 16 DWORDs (64B row)
                | (3u << 25));    // pad_amount: 4 DWORDs (16B)
  g1[1] = (int)((td0 & 0xffffu) << 16);                          // dim0[15:0]
  g1[2] = (int)(((td0 >> 16) & 0xffffu) | ((td1 & 0xffffu) << 16));
  g1[3] = (int)(((td1 >> 16) & 0xffffu) | (32u << 16));          // tile_dim0=32
  g1[4] = (int)(tileRows & 0xffffu);                             // tile_dim1
  g1[5] = (int)strideElems;                                      // dim0_stride lo
  g1[6] = 0;                                                     // stride hi, dim1_stride lo
  g1[7] = 0;
  v4i z4 = {0, 0, 0, 0};
#if __clang_major__ >= 23
  v8i z8 = {0, 0, 0, 0, 0, 0, 0, 0};
  __builtin_amdgcn_tensor_load_to_lds(g0, g1, z4, z4, z8, 0);
#else
  __builtin_amdgcn_tensor_load_to_lds(g0, g1, z4, z4, 0);
#endif
}
#endif  // USE_TDM

// ---------------------------------------------------------------------------
// fp32 -> f16 cast
// ---------------------------------------------------------------------------
__global__ void cast_f32_to_f16(const float* __restrict__ src,
                                _Float16* __restrict__ dst, size_t n) {
  size_t i = (size_t)blockIdx.x * blockDim.x + threadIdx.x;
  size_t stride = (size_t)gridDim.x * blockDim.x;
  for (; i < n; i += stride) dst[i] = (_Float16)src[i];
}

// ---------------------------------------------------------------------------
// GEMM: C[m,n] = sum_k A[m,k] * Bw[n,k] + bias[n]
//   A:  [M,K] f16 row-major,  Bw: [N,K] f16 row-major (torch weight layout)
// Block tile 128x256, 8 waves (2 in M x 4 in N), wave tile 64x64 (16 WMMAs /
// k-step). K-tiles of 32 staged through double-buffered LDS by the TDM.
// M % 128 == 0, N % 256 == 0, K % 32 == 0.
// ---------------------------------------------------------------------------
template <typename OutT>
__global__ __launch_bounds__(256) void gemm_nt_wmma(
    const _Float16* __restrict__ A, const _Float16* __restrict__ Bw,
    const float* __restrict__ bias, OutT* __restrict__ C,
    int M, int N, int K) {
  constexpr int LDST  = 40;            // LDS row stride in halves (32 + 8 pad)
  constexpr int ATILE = 128 * LDST;    // halves per A buffer
  constexpr int BTILE = 256 * LDST;    // halves per B buffer
  __shared__ _Float16 sA[2 * ATILE];
  __shared__ _Float16 sB[2 * BTILE];   // total LDS: 61440 B

  const int tid  = threadIdx.x;
  const int wave = tid >> 5;
  const int lane = tid & 31;
  const int wm   = wave >> 2;          // 0..1
  const int wn   = wave & 3;           // 0..3
  const int m0   = blockIdx.y * 128;
  const int n0   = blockIdx.x * 256;
  const int KT   = K >> 5;

  v8f acc[4][4] = {};

  const int aRow = tid >> 1;           // 0..127, 16 halves each
  const int aOff = (tid & 1) * 16;
  const int bRow = tid;                // 0..255, 32 halves each

  // ---- stage k-tile 0 ----
#if USE_TDM
  if (wave == 0) {
    tdm_load_tile32(A + (size_t)m0 * K, (unsigned)(unsigned long long)(sA), 128, K);
    tdm_load_tile32(Bw + (size_t)n0 * K, (unsigned)(unsigned long long)(sB), 256, K);
    __builtin_amdgcn_s_wait_tensorcnt(0);
  }
#else
  {
    const uint4* asrc = (const uint4*)(A + (size_t)(m0 + aRow) * K + aOff);
    uint4* adst = (uint4*)(sA + aRow * LDST + aOff);
    adst[0] = asrc[0]; adst[1] = asrc[1];
    const uint4* bsrc = (const uint4*)(Bw + (size_t)(n0 + bRow) * K);
    uint4* bdst = (uint4*)(sB + bRow * LDST);
    bdst[0] = bsrc[0]; bdst[1] = bsrc[1]; bdst[2] = bsrc[2]; bdst[3] = bsrc[3];
  }
#endif
  __syncthreads();

  for (int kt = 0; kt < KT; ++kt) {
    const int cur = kt & 1;
    const int nxt = cur ^ 1;
    const bool havePrefetch = (kt + 1 < KT);
    const int kn = (kt + 1) << 5;

    // ---- kick off next tile (overlaps with the WMMAs below) ----
#if USE_TDM
    if (wave == 0 && havePrefetch) {
      tdm_load_tile32(A + (size_t)m0 * K + kn,
                      (unsigned)(unsigned long long)(sA + nxt * ATILE), 128, K);
      tdm_load_tile32(Bw + (size_t)n0 * K + kn,
                      (unsigned)(unsigned long long)(sB + nxt * BTILE), 256, K);
    }
#else
    uint4 ra0 = {}, ra1 = {}, rb0 = {}, rb1 = {}, rb2 = {}, rb3 = {};
    if (havePrefetch) {
      const uint4* asrc = (const uint4*)(A + (size_t)(m0 + aRow) * K + kn + aOff);
      ra0 = asrc[0]; ra1 = asrc[1];
      const uint4* bsrc = (const uint4*)(Bw + (size_t)(n0 + bRow) * K + kn);
      rb0 = bsrc[0]; rb1 = bsrc[1]; rb2 = bsrc[2]; rb3 = bsrc[3];
    }
#endif

    // ---- compute on current tile: 16 WMMAs per wave ----
    const _Float16* tA = sA + cur * ATILE;
    const _Float16* tB = sB + cur * BTILE;
    v16h af[4], bf[4];
#pragma unroll
    for (int i = 0; i < 4; ++i)
      af[i] = load_frag16(tA + (wm * 64 + i * 16) * LDST, LDST);
#pragma unroll
    for (int j = 0; j < 4; ++j)
      bf[j] = load_frag16(tB + (wn * 64 + j * 16) * LDST, LDST);
#pragma unroll
    for (int i = 0; i < 4; ++i)
#pragma unroll
      for (int j = 0; j < 4; ++j)
        acc[i][j] = wmma_f16(af[i], bf[j], acc[i][j]);

    // ---- land the prefetch, then rotate buffers ----
#if USE_TDM
    if (wave == 0 && havePrefetch) __builtin_amdgcn_s_wait_tensorcnt(0);
#else
    if (havePrefetch) {
      uint4* adst = (uint4*)(sA + nxt * ATILE + aRow * LDST + aOff);
      adst[0] = ra0; adst[1] = ra1;
      uint4* bdst = (uint4*)(sB + nxt * BTILE + bRow * LDST);
      bdst[0] = rb0; bdst[1] = rb1; bdst[2] = rb2; bdst[3] = rb3;
    }
#endif
    __syncthreads();
  }

  // ---- epilogue: C/D layout -> vgpr r holds row r / r+8 per half-wave ----
  const int halfOff = (lane < 16) ? 0 : 8;
#pragma unroll
  for (int j = 0; j < 4; ++j) {
    const int col = n0 + wn * 64 + j * 16 + (lane & 15);
    const float bv = bias[col];
#pragma unroll
    for (int i = 0; i < 4; ++i) {
      const int rowBase = m0 + wm * 64 + i * 16 + halfOff;
#pragma unroll
      for (int r = 0; r < 8; ++r)
        C[(size_t)(rowBase + r) * N + col] = (OutT)(acc[i][j][r] + bv);
    }
  }
}

// ---------------------------------------------------------------------------
// Flash-style causal attention.
//   qkv: [B*T, 3*D_MODEL] f16 (q | k | v, heads interleaved in D_MODEL)
//   yh : [B*T, D_MODEL]   f16 output (pre-projection)
// Grid: x = query block (128 rows), y = b*N_HEAD + h. Block = 256 thr (8 waves).
// ---------------------------------------------------------------------------
__global__ __launch_bounds__(256) void attn_causal_wmma(
    const _Float16* __restrict__ qkv, _Float16* __restrict__ yh) {
  constexpr int C3  = 3 * D_MODEL;
  constexpr int CK  = D_MODEL;
  constexpr int CV  = 2 * D_MODEL;
  constexpr int LD  = 72;                  // LDS row stride (64 + 8 pad halves)

  __shared__ _Float16 sQ[128 * LD];        // 128 x 64 queries
  __shared__ _Float16 sK[64 * LD];         // 64  x 64 keys        [k][d]
  __shared__ _Float16 sVT[64 * LD];        // 64  x 64 values^T    [d][k]
  __shared__ _Float16 sP[8 * 16 * LD];     // per-wave P tile 16x64

  const int tid  = threadIdx.x;
  const int wave = tid >> 5;
  const int lane = tid & 31;
  const int bh   = blockIdx.y;
  const int b    = bh >> 4;
  const int h    = bh & (N_HEAD - 1);
  const int q0   = blockIdx.x * 128;
  const size_t rowBaseG = (size_t)b * SEQ_T;

  // ---- stage Q block (each thread: 32 halves) ----
  {
    const int row = tid >> 1;
    const int off = (tid & 1) * 32;
    const uint4* src =
        (const uint4*)(qkv + (rowBaseG + q0 + row) * C3 + h * D_HEAD + off);
    uint4* dst = (uint4*)(sQ + row * LD + off);
    dst[0] = src[0]; dst[1] = src[1]; dst[2] = src[2]; dst[3] = src[3];
  }
  __syncthreads();

  v16h aQ[2];
  aQ[0] = load_frag16(sQ + (wave * 16) * LD + 0, LD);
  aQ[1] = load_frag16(sQ + (wave * 16) * LD + 32, LD);

  v8f o[4] = {};
  float mrow[8], lrow[8];
#pragma unroll
  for (int r = 0; r < 8; ++r) { mrow[r] = -1.0e30f; lrow[r] = 0.0f; }

  const float NEG_INF = -__builtin_inff();
  const float scale = 0.125f;  // 1/sqrt(D_HEAD)
  const int halfOff = (lane < 16) ? 0 : 8;
  const int qRowBase = q0 + wave * 16 + halfOff;  // + r -> global query row

  const int ktiles = 2 * blockIdx.x + 2;  // only tiles touching causal region
  for (int kt = 0; kt < ktiles; ++kt) {
    const int k0 = kt * 64;
    __syncthreads();  // previous iteration done with sK/sVT
    {  // stage K tile 64x64 (each thread: 16 halves)
      const int row = tid >> 2;
      const int off = (tid & 3) * 16;
      const uint4* src =
          (const uint4*)(qkv + (rowBaseG + k0 + row) * C3 + CK + h * D_HEAD + off);
      uint4* dst = (uint4*)(sK + row * LD + off);
      dst[0] = src[0]; dst[1] = src[1];
    }
    {  // stage V tile transposed: sVT[d][k]
      const int row  = tid >> 2;
      const int doff = (tid & 3) * 16;
      const _Float16* src =
          qkv + (rowBaseG + k0 + row) * C3 + CV + h * D_HEAD + doff;
      v8h v0 = *(const v8h*)(src);
      v8h v1 = *(const v8h*)(src + 8);
#pragma unroll
      for (int j = 0; j < 8; ++j) {
        sVT[(doff + j) * LD + row]     = v0[j];
        sVT[(doff + 8 + j) * LD + row] = v1[j];
      }
    }
    // ---- prefetch next K/V tile into WGP$/L2 (global_prefetch_b8);
    //      overlaps with the WMMA/softmax work on the current tile ----
    if (kt + 1 < ktiles) {
      const size_t prow = rowBaseG + (size_t)(k0 + 64) + (tid >> 2);
      __builtin_prefetch(qkv + prow * C3 + CK + h * D_HEAD, 0, 3);
      __builtin_prefetch(qkv + prow * C3 + CV + h * D_HEAD, 0, 3);
    }
    __syncthreads();

    // ---- S = Q K^T (16 x 64 per wave) ----
    v8f s[4];
#pragma unroll
    for (int nt = 0; nt < 4; ++nt) {
      v16h bf0 = load_frag16(sK + (nt * 16) * LD + 0, LD);
      v16h bf1 = load_frag16(sK + (nt * 16) * LD + 32, LD);
      v8f c = {};
      c = wmma_f16(aQ[0], bf0, c);
      c = wmma_f16(aQ[1], bf1, c);
      s[nt] = c;
    }

    // ---- online softmax ----
    const int col0 = k0 + (lane & 15);
    float mnew[8];
#pragma unroll
    for (int r = 0; r < 8; ++r) {
      const int qrow = qRowBase + r;
      float mx = mrow[r];
#pragma unroll
      for (int nt = 0; nt < 4; ++nt) {
        float v = s[nt][r] * scale;
        v = ((col0 + nt * 16) <= qrow) ? v : NEG_INF;
        s[nt][r] = v;
        mx = fmaxf(mx, v);
      }
#pragma unroll
      for (int m = 1; m < 16; m <<= 1) mx = fmaxf(mx, __shfl_xor(mx, m, 32));
      mnew[r] = mx;
    }

    _Float16* pBase = sP + wave * 16 * LD;
#pragma unroll
    for (int r = 0; r < 8; ++r) {
      const float es = __expf(mrow[r] - mnew[r]);
      const int prow = r + halfOff;
      float sum = 0.0f;
#pragma unroll
      for (int nt = 0; nt < 4; ++nt) {
        const float p = __expf(s[nt][r] - mnew[r]);
        sum += p;
        pBase[prow * LD + nt * 16 + (lane & 15)] = (_Float16)p;
      }
#pragma unroll
      for (int m = 1; m < 16; m <<= 1) sum += __shfl_xor(sum, m, 32);
      lrow[r] = lrow[r] * es + sum;
      mrow[r] = mnew[r];
#pragma unroll
      for (int d = 0; d < 4; ++d) o[d][r] = o[d][r] * es;
    }

    // wave-private LDS round-trip: DS ops are in-order per wave; make the
    // compiler respect it and drain the DS counter explicitly.
    asm volatile("s_wait_dscnt 0" ::: "memory");

    // ---- O += P V ----
    v16h aP0 = load_frag16(pBase + 0, LD);
    v16h aP1 = load_frag16(pBase + 32, LD);
#pragma unroll
    for (int d = 0; d < 4; ++d) {
      v16h bf0 = load_frag16(sVT + (d * 16) * LD + 0, LD);
      v16h bf1 = load_frag16(sVT + (d * 16) * LD + 32, LD);
      o[d] = wmma_f16(aP0, bf0, o[d]);
      o[d] = wmma_f16(aP1, bf1, o[d]);
    }
  }

  // ---- normalize and write yh (f16) ----
  const int colBase = h * D_HEAD + (lane & 15);
#pragma unroll
  for (int r = 0; r < 8; ++r) {
    const float inv = 1.0f / lrow[r];
    const size_t row = rowBaseG + qRowBase + r;
#pragma unroll
    for (int d = 0; d < 4; ++d)
      yh[row * D_MODEL + colBase + d * 16] = (_Float16)(o[d][r] * inv);
  }
}

// ---------------------------------------------------------------------------
// launch
// ---------------------------------------------------------------------------
extern "C" void kernel_launch(void* const* d_in, const int* in_sizes, int n_in,
                              void* d_out, int out_size, void* d_ws, size_t ws_size,
                              hipStream_t stream) {
  const float* x      = (const float*)d_in[0];
  const float* w_qkv  = (const float*)d_in[1];
  const float* b_qkv  = (const float*)d_in[2];
  const float* w_proj = (const float*)d_in[3];
  const float* b_proj = (const float*)d_in[4];
  float* out = (float*)d_out;

  // f16 workspace carve-up
  _Float16* ws = (_Float16*)d_ws;
  _Float16* xh     = ws;
  _Float16* wqkvh  = xh     + (size_t)MROWS * D_MODEL;          //  8192x1024
  _Float16* wprojh = wqkvh  + (size_t)3 * D_MODEL * D_MODEL;    //  3072x1024
  _Float16* qkvh   = wprojh + (size_t)D_MODEL * D_MODEL;        //  1024x1024
  _Float16* yh     = qkvh   + (size_t)MROWS * 3 * D_MODEL;      //  8192x3072

  const int CAST_BLK = 256;
  {
    size_t n = (size_t)MROWS * D_MODEL;
    cast_f32_to_f16<<<dim3((n + CAST_BLK - 1) / CAST_BLK), CAST_BLK, 0, stream>>>(x, xh, n);
  }
  {
    size_t n = (size_t)3 * D_MODEL * D_MODEL;
    cast_f32_to_f16<<<dim3((n + CAST_BLK - 1) / CAST_BLK), CAST_BLK, 0, stream>>>(w_qkv, wqkvh, n);
  }
  {
    size_t n = (size_t)D_MODEL * D_MODEL;
    cast_f32_to_f16<<<dim3((n + CAST_BLK - 1) / CAST_BLK), CAST_BLK, 0, stream>>>(w_proj, wprojh, n);
  }

  // qkv = x @ w_qkv^T + b_qkv   -> f16 [8192, 3072]
  gemm_nt_wmma<_Float16><<<dim3(3 * D_MODEL / 256, MROWS / 128), 256, 0, stream>>>(
      xh, wqkvh, b_qkv, qkvh, MROWS, 3 * D_MODEL, D_MODEL);

  // attention -> yh f16 [8192, 1024]
  attn_causal_wmma<<<dim3(SEQ_T / 128, BATCH * N_HEAD), 256, 0, stream>>>(qkvh, yh);

  // out = yh @ w_proj^T + b_proj -> f32 [8192, 1024]
  gemm_nt_wmma<float><<<dim3(D_MODEL / 256, MROWS / 128), 256, 0, stream>>>(
      yh, wprojh, b_proj, out, MROWS, D_MODEL, D_MODEL);
}
